// TransformerDecoderLayer_37297495998529
// MI455X (gfx1250) — matrix operand order repaired and checked by reading, not compiled, not run on hardware
//
#include <hip/hip_runtime.h>
#include <math.h>

typedef __bf16 bf16;
typedef __attribute__((ext_vector_type(8)))  bf16  bf16x8;
typedef __attribute__((ext_vector_type(16))) bf16  bf16x16;
typedef __attribute__((ext_vector_type(8)))  float f32x8;

#define DEVI __device__ __forceinline__

DEVI bf16 f2bf(float x) {
  unsigned u = __float_as_uint(x);
  unsigned r = (u + 0x7fffu + ((u >> 16) & 1u)) >> 16;   // round-to-nearest-even
  unsigned short s = (unsigned short)r;
  bf16 b; __builtin_memcpy(&b, &s, 2); return b;
}

DEVI bf16x16 cat8(bf16x8 c0, bf16x8 c1) {
  bf16x16 a;
#pragma unroll
  for (int i = 0; i < 8; i++) { a[i] = c0[i]; a[8 + i] = c1[i]; }
  return a;
}

// A-operand (16xK tile): lane (r,h): halfs[0..7]=row r, K = k0+8h.. ; halfs[8..15]= K = k0+16+8h..
DEVI bf16x16 load_a_frag(const bf16* base, int ld, int row0, int k0, int lane) {
  int r = lane & 15, h = lane >> 4;
  const bf16* p = base + (size_t)(row0 + r) * ld + k0;
  bf16x8 c0 = *(const bf16x8*)(p + 8 * h);
  bf16x8 c1 = *(const bf16x8*)(p + 16 + 8 * h);
  return cat8(c0, c1);
}

// B-operand (Kx16 tile), source stored row-major with rows = B-columns:
// lane (r,h): column col0+r, halfs[0..7] = K k0+8h.. ; halfs[8..15] = K k0+16+8h..
DEVI bf16x16 load_b_frag(const bf16* base, int ld, int col0, int k0, int lane) {
  int r = lane & 15, h = lane >> 4;
  const bf16* p = base + (size_t)(col0 + r) * ld + k0;
  bf16x8 c0 = *(const bf16x8*)(p + 8 * h);
  bf16x8 c1 = *(const bf16x8*)(p + 16 + 8 * h);
  return cat8(c0, c1);
}

DEVI f32x8 wmma_bf(bf16x16 a, bf16x16 b, f32x8 c) {
  return __builtin_amdgcn_wmma_f32_16x16x32_bf16(false, a, false, b, (short)0, c, false, false);
}

// ---------------------------------------------------------------------------
// GEMM: out[M,N] = A[M,K](bf16) * W[N,K]^T(bf16) + bias[N] (+Cin) (ReLU opt)
// 128 threads = 4 waves; wave -> 32x32 tile; workgroup -> 64x64 tile.
// ---------------------------------------------------------------------------
__global__ __launch_bounds__(128)
void gemm_wmma(const bf16* __restrict__ A, const bf16* __restrict__ W,
               const float* __restrict__ bias, const float* __restrict__ Cin,
               float* __restrict__ out32, bf16* __restrict__ out16,
               int M, int N, int K, int relu) {
  int lane = threadIdx.x & 31;
  int wave = threadIdx.x >> 5;
  int m0 = blockIdx.y * 64 + (wave >> 1) * 32;
  int n0 = blockIdx.x * 64 + (wave & 1) * 32;

  f32x8 acc[2][2] = {};
  for (int k = 0; k < K; k += 32) {
    bf16x16 a0 = load_a_frag(A, K, m0,      k, lane);
    bf16x16 a1 = load_a_frag(A, K, m0 + 16, k, lane);
    bf16x16 b0 = load_b_frag(W, K, n0,      k, lane);
    bf16x16 b1 = load_b_frag(W, K, n0 + 16, k, lane);
    acc[0][0] = wmma_bf(a0, b0, acc[0][0]);
    acc[0][1] = wmma_bf(a0, b1, acc[0][1]);
    acc[1][0] = wmma_bf(a1, b0, acc[1][0]);
    acc[1][1] = wmma_bf(a1, b1, acc[1][1]);
  }
  int r = lane & 15, hh = lane >> 4;
#pragma unroll
  for (int i = 0; i < 2; i++)
#pragma unroll
    for (int j = 0; j < 2; j++) {
      int col = n0 + 16 * j + r;
      float bcol = bias ? bias[col] : 0.f;
#pragma unroll
      for (int e = 0; e < 8; e++) {
        int row = m0 + 16 * i + e + 8 * hh;
        size_t idx = (size_t)row * N + col;
        float v = acc[i][j][e] + bcol;
        if (Cin) v += Cin[idx];
        if (relu) v = fmaxf(v, 0.f);
        if (out32) out32[idx] = v;
        if (out16) out16[idx] = f2bf(v);
      }
    }
}

// ---------------------------------------------------------------------------
// Flash attention. Qp: [BH,L,HDQK] (pre-scaled), Kp: [BH,S,HDQK],
// Vt: [BH,32,S] (transposed), out: [L,B,256] f32. One wave = 16 queries.
// ---------------------------------------------------------------------------
template <int HDQK>
__global__ __launch_bounds__(128)
void attn_wmma(const bf16* __restrict__ Qp, const bf16* __restrict__ Kp,
               const bf16* __restrict__ Vt, float* __restrict__ out,
               int L, int S, int B, int H) {
  __shared__ __align__(16) bf16 psh[4][16][32];
  int lane = threadIdx.x & 31;
  int wave = threadIdx.x >> 5;
  int bh = blockIdx.y;
  int b = bh / H, h = bh % H;
  int l0 = blockIdx.x * 64 + wave * 16;

  const bf16* Q  = Qp + (size_t)bh * L * HDQK;
  const bf16* Km = Kp + (size_t)bh * S * HDQK;
  const bf16* V  = Vt + (size_t)bh * 32 * S;

  bf16x16 qf[HDQK / 32];
#pragma unroll
  for (int c = 0; c < HDQK / 32; c++) qf[c] = load_a_frag(Q, HDQK, l0, 32 * c, lane);

  f32x8 o0 = {}, o1 = {};
  float mi[8], li[8];
#pragma unroll
  for (int e = 0; e < 8; e++) { mi[e] = -INFINITY; li[e] = 0.f; }

  int r = lane & 15, hh = lane >> 4;
  bf16* P = &psh[wave][0][0];

  for (int kb = 0; kb < S; kb += 32) {
    f32x8 s0 = {}, s1 = {};
#pragma unroll
    for (int c = 0; c < HDQK / 32; c++) {
      bf16x16 k0f = load_b_frag(Km, HDQK, kb,      32 * c, lane);
      bf16x16 k1f = load_b_frag(Km, HDQK, kb + 16, 32 * c, lane);
      s0 = wmma_bf(qf[c], k0f, s0);
      s1 = wmma_bf(qf[c], k1f, s1);
    }
    // online softmax per row (row = e + 8*hh); cols live across 16-lane groups
#pragma unroll
    for (int e = 0; e < 8; e++) {
      float mx = fmaxf(s0[e], s1[e]);
#pragma unroll
      for (int off = 8; off > 0; off >>= 1) mx = fmaxf(mx, __shfl_xor(mx, off, 16));
      float mn  = fmaxf(mi[e], mx);
      float scl = __expf(mi[e] - mn);
      float e0  = __expf(s0[e] - mn);
      float e1  = __expf(s1[e] - mn);
      float sm  = e0 + e1;
#pragma unroll
      for (int off = 8; off > 0; off >>= 1) sm += __shfl_xor(sm, off, 16);
      li[e] = li[e] * scl + sm;
      mi[e] = mn;
      o0[e] *= scl; o1[e] *= scl;
      P[(e + 8 * hh) * 32 + r]      = f2bf(e0);
      P[(e + 8 * hh) * 32 + 16 + r] = f2bf(e1);
    }
    asm volatile("s_wait_dscnt 0" ::: "memory");
    // P (C layout in LDS) -> A-operand fragment
    bf16x8 pc0 = *(const bf16x8*)(P + r * 32 + 8 * hh);
    bf16x8 pc1 = *(const bf16x8*)(P + r * 32 + 16 + 8 * hh);
    bf16x16 pf = cat8(pc0, pc1);
    // V (transposed [vd][S]) -> B-operand fragments, contiguous loads
    bf16x16 v0f = load_b_frag(V, S, 0,  kb, lane);
    bf16x16 v1f = load_b_frag(V, S, 16, kb, lane);
    o0 = wmma_bf(pf, v0f, o0);
    o1 = wmma_bf(pf, v1f, o1);
  }
#pragma unroll
  for (int e = 0; e < 8; e++) {
    float inv = 1.f / li[e];
    int row = l0 + e + 8 * hh;
    size_t base = (size_t)row * B * 256 + (size_t)b * 256 + h * 32;
    out[base + r]      = o0[e] * inv;
    out[base + 16 + r] = o1[e] * inv;
  }
}

// ---------------------------------------------------------------------------
// LayerNorm(x + resid) over rows of 256; one wave per row.
// ---------------------------------------------------------------------------
__global__ __launch_bounds__(256)
void ln_res(const float* __restrict__ x, const float* __restrict__ resid,
            const float* __restrict__ g, const float* __restrict__ bb,
            float* __restrict__ out32, bf16* __restrict__ out16, int rows) {
  int lane = threadIdx.x & 31;
  int wave = threadIdx.x >> 5;
  int row = blockIdx.x * 8 + wave;
  if (row >= rows) return;
  size_t base = (size_t)row * 256;
  float v[8], s = 0.f, ss = 0.f;
#pragma unroll
  for (int i = 0; i < 8; i++) {
    int c = lane * 8 + i;
    float t = x[base + c] + (resid ? resid[base + c] : 0.f);
    v[i] = t; s += t; ss += t * t;
  }
#pragma unroll
  for (int off = 16; off > 0; off >>= 1) {
    s  += __shfl_xor(s,  off, 32);
    ss += __shfl_xor(ss, off, 32);
  }
  float mean = s * (1.f / 256.f);
  float var  = ss * (1.f / 256.f) - mean * mean;
  float rstd = rsqrtf(var + 1e-5f);
#pragma unroll
  for (int i = 0; i < 8; i++) {
    int c = lane * 8 + i;
    float y = (v[i] - mean) * rstd * g[c] + bb[c];
    if (out32) out32[base + c] = y;
    if (out16) out16[base + c] = f2bf(y);
  }
}

__global__ void cvt_bf16(const float* __restrict__ in, bf16* __restrict__ out, size_t n) {
  size_t i = (size_t)blockIdx.x * 256 + threadIdx.x;
  if (i < n) out[i] = f2bf(in[i]);
}

// in [T,B,256] f32 -> out [B,8,T,32] bf16, scaled
__global__ void pack_qk(const float* __restrict__ in, bf16* __restrict__ out,
                        int T, int B, float scale, size_t n) {
  size_t i = (size_t)blockIdx.x * 256 + threadIdx.x;
  if (i >= n) return;
  int d = (int)(i & 31);
  int t = (int)((i >> 5) % T);
  size_t q = i / ((size_t)32 * T);
  int h = (int)(q & 7), b = (int)(q >> 3);
  out[i] = f2bf(in[((size_t)t * B + b) * 256 + h * 32 + d] * scale);
}

// in [T,B,256] f32 -> out [B,8,32,T] bf16 (transposed V)
__global__ void pack_vt(const float* __restrict__ in, bf16* __restrict__ out,
                        int T, int B, size_t n) {
  size_t i = (size_t)blockIdx.x * 256 + threadIdx.x;
  if (i >= n) return;
  int t = (int)(i % T);
  size_t q = i / T;
  int d = (int)(q & 31); q >>= 5;
  int h = (int)(q & 7),  b = (int)(q >> 3);
  out[i] = f2bf(in[((size_t)t * B + b) * 256 + h * 32 + d]);
}

// out [B,8,T,64]: d<32 -> (a [+ aAdd]); d>=32 -> c2 ; all * scale
__global__ void pack_cat64(const float* __restrict__ a, const float* __restrict__ aAdd,
                           const float* __restrict__ c2, bf16* __restrict__ out,
                           int T, int B, float scale, size_t n) {
  size_t i = (size_t)blockIdx.x * 256 + threadIdx.x;
  if (i >= n) return;
  int d = (int)(i & 63);
  int t = (int)((i >> 6) % T);
  size_t q = i / ((size_t)64 * T);
  int h = (int)(q & 7), b = (int)(q >> 3);
  size_t src = ((size_t)t * B + b) * 256 + h * 32 + (d & 31);
  float v = (d < 32) ? (a[src] + (aAdd ? aAdd[src] : 0.f)) : c2[src];
  out[i] = f2bf(v * scale);
}

extern "C" void kernel_launch(void* const* d_in, const int* in_sizes, int n_in,
                              void* d_out, int out_size, void* d_ws, size_t ws_size,
                              hipStream_t stream) {
  (void)in_sizes; (void)n_in; (void)out_size; (void)ws_size;
  const int D = 256, H = 8, FF = 2048, NQ = 512, B = 16, S = 2048;
  const int MQ = NQ * B, MS = S * B;

  const float* tgt   = (const float*)d_in[0];
  const float* mem   = (const float*)d_in[1];
  const float* pos   = (const float*)d_in[2];
  const float* qpos  = (const float*)d_in[3];
  const float* qsine = (const float*)d_in[4];
  const float* W_[13]; const float* Bb_[13];
  for (int i = 0; i < 13; i++) { W_[i] = (const float*)d_in[5 + 2 * i]; Bb_[i] = (const float*)d_in[6 + 2 * i]; }
  // 0 sa_qc,1 sa_qp,2 sa_kc,3 sa_kp,4 sa_v,5 ca_qc,6 ca_qp,7 ca_kc,8 ca_kp,9 ca_v,10 ca_qps,11 l1,12 l2
  const float* g1 = (const float*)d_in[31]; const float* b1 = (const float*)d_in[32];
  const float* g2 = (const float*)d_in[33]; const float* b2 = (const float*)d_in[34];
  const float* g3 = (const float*)d_in[35]; const float* b3 = (const float*)d_in[36];

  char* wsb = (char*)d_ws; size_t off = 0;
  auto alloc = [&](size_t bytes) -> void* {
    void* p = wsb + off; off = (off + bytes + 255) & ~(size_t)255; return p;
  };

  bf16* tgt16 = (bf16*)alloc((size_t)MQ * D * 2);
  bf16* qp16  = (bf16*)alloc((size_t)MQ * D * 2);
  bf16* qse16 = (bf16*)alloc((size_t)MQ * D * 2);
  bf16* mem16 = (bf16*)alloc((size_t)MS * D * 2);
  bf16* pos16 = (bf16*)alloc((size_t)MS * D * 2);

  bf16* wb[13]; size_t wsz[13];
  for (int i = 0; i < 13; i++) {
    wsz[i] = (i < 11) ? (size_t)D * D : (size_t)D * FF;
    wb[i] = (bf16*)alloc(wsz[i] * 2);
  }

  float* q_sa  = (float*)alloc((size_t)MQ * D * 4);
  float* k_sa  = (float*)alloc((size_t)MQ * D * 4);
  float* v_sa  = (float*)alloc((size_t)MQ * D * 4);
  float* attnO = (float*)alloc((size_t)MQ * D * 4);
  float* tgt1  = (float*)alloc((size_t)MQ * D * 4);
  bf16*  tgt1_16 = (bf16*)alloc((size_t)MQ * D * 2);
  float* q_ca  = (float*)alloc((size_t)MQ * D * 4);
  float* qse_p = (float*)alloc((size_t)MQ * D * 4);
  float* kc    = (float*)alloc((size_t)MS * D * 4);
  float* kp    = (float*)alloc((size_t)MS * D * 4);
  float* vca   = (float*)alloc((size_t)MS * D * 4);
  float* tgt2  = (float*)alloc((size_t)MQ * D * 4);
  bf16*  tgt2_16 = (bf16*)alloc((size_t)MQ * D * 2);
  bf16*  ff1_16  = (bf16*)alloc((size_t)MQ * FF * 2);
  float* ff2   = (float*)alloc((size_t)MQ * D * 4);

  bf16* Qsa  = (bf16*)alloc((size_t)B * H * NQ * 32 * 2);
  bf16* Ksa  = (bf16*)alloc((size_t)B * H * NQ * 32 * 2);
  bf16* Vsa  = (bf16*)alloc((size_t)B * H * NQ * 32 * 2);
  bf16* Qcat = (bf16*)alloc((size_t)B * H * NQ * 64 * 2);
  bf16* Kcat = (bf16*)alloc((size_t)B * H * S  * 64 * 2);
  bf16* Vca  = (bf16*)alloc((size_t)B * H * S  * 32 * 2);

  auto cvt = [&](const float* in, bf16* o, size_t n) {
    cvt_bf16<<<dim3((unsigned)((n + 255) / 256)), 256, 0, stream>>>(in, o, n);
  };
  cvt(tgt, tgt16, (size_t)MQ * D);
  cvt(qpos, qp16, (size_t)MQ * D);
  cvt(qsine, qse16, (size_t)MQ * D);
  cvt(mem, mem16, (size_t)MS * D);
  cvt(pos, pos16, (size_t)MS * D);
  for (int i = 0; i < 13; i++) cvt(W_[i], wb[i], wsz[i]);

  auto gemm = [&](const bf16* A, int iw, const float* Cin, float* o32, bf16* o16,
                  int M, int N, int K, int relu) {
    gemm_wmma<<<dim3(N / 64, M / 64), 128, 0, stream>>>(A, wb[iw], Bb_[iw], Cin, o32, o16, M, N, K, relu);
  };

  // ---- self-attention projections ----
  gemm(tgt16, 0, nullptr, q_sa, nullptr, MQ, D, D, 0);
  gemm(qp16,  1, q_sa,    q_sa, nullptr, MQ, D, D, 0);
  gemm(tgt16, 2, nullptr, k_sa, nullptr, MQ, D, D, 0);
  gemm(qp16,  3, k_sa,    k_sa, nullptr, MQ, D, D, 0);
  gemm(tgt16, 4, nullptr, v_sa, nullptr, MQ, D, D, 0);

  size_t nq32 = (size_t)B * H * NQ * 32;
  float sa_scale = 1.f / sqrtf(32.f);
  pack_qk<<<dim3((unsigned)((nq32 + 255) / 256)), 256, 0, stream>>>(q_sa, Qsa, NQ, B, sa_scale, nq32);
  pack_qk<<<dim3((unsigned)((nq32 + 255) / 256)), 256, 0, stream>>>(k_sa, Ksa, NQ, B, 1.f, nq32);
  pack_vt<<<dim3((unsigned)((nq32 + 255) / 256)), 256, 0, stream>>>(v_sa, Vsa, NQ, B, nq32);

  attn_wmma<32><<<dim3(NQ / 64, B * H), 128, 0, stream>>>(Qsa, Ksa, Vsa, attnO, NQ, NQ, B, H);
  ln_res<<<dim3(MQ / 8), 256, 0, stream>>>(tgt, attnO, g1, b1, tgt1, tgt1_16, MQ);

  // ---- cross-attention projections ----
  gemm(tgt1_16, 5, nullptr, q_ca,  nullptr, MQ, D, D, 0);
  gemm(qp16,    6, q_ca,    q_ca,  nullptr, MQ, D, D, 0);
  gemm(mem16,   7, nullptr, kc,    nullptr, MS, D, D, 0);
  gemm(pos16,   8, nullptr, kp,    nullptr, MS, D, D, 0);
  gemm(mem16,   9, nullptr, vca,   nullptr, MS, D, D, 0);
  gemm(qse16,  10, nullptr, qse_p, nullptr, MQ, D, D, 0);

  size_t nqc = (size_t)B * H * NQ * 64;
  size_t nkc = (size_t)B * H * S * 64;
  size_t nvc = (size_t)B * H * S * 32;
  pack_cat64<<<dim3((unsigned)((nqc + 255) / 256)), 256, 0, stream>>>(q_ca, nullptr, qse_p, Qcat, NQ, B, 0.125f, nqc);
  pack_cat64<<<dim3((unsigned)((nkc + 255) / 256)), 256, 0, stream>>>(kc, kp, kp, Kcat, S, B, 1.f, nkc);
  pack_vt<<<dim3((unsigned)((nvc + 255) / 256)), 256, 0, stream>>>(vca, Vca, S, B, nvc);

  attn_wmma<64><<<dim3(NQ / 64, B * H), 128, 0, stream>>>(Qcat, Kcat, Vca, attnO, NQ, S, B, H);
  ln_res<<<dim3(MQ / 8), 256, 0, stream>>>(tgt1, attnO, g2, b2, tgt2, tgt2_16, MQ);

  // ---- FFN ----
  gemm(tgt2_16, 11, nullptr, nullptr, ff1_16, MQ, FF, D, 1);  // relu fused, bf16 out
  gemm(ff1_16,  12, nullptr, ff2, nullptr, MQ, D, FF, 0);
  ln_res<<<dim3(MQ / 8), 256, 0, stream>>>(tgt2, ff2, g3, b3, (float*)d_out, nullptr, MQ);
}
